// MMsingle_attention_23699629539653
// MI455X (gfx1250) — compile-verified, matrix operand order checked by reading, not run on hardware
//
#include <hip/hip_runtime.h>
#include <hip/hip_bf16.h>
#include <math.h>
#include <stdint.h>

typedef __attribute__((ext_vector_type(16))) _Float16 v16h;
typedef __attribute__((ext_vector_type(8)))  float    v8f;

#define HCNT 16     // heads
#define DH   128    // head dim
#define BQ   64     // q rows per block
#define BK   64     // keys per block

// Load a 16-bit WMMA fragment (16 halves per lane) from a row-major LDS tile.
// Caller passes p = &tile[row*ld + kbase + half*8]; layout needs halves
// [p..p+7] (K lo) and [p+16..p+23] (K hi) per the ISA 16-bit A layout.
__device__ __forceinline__ v16h load_frag(const _Float16* p) {
    union { v16h v; int4 q[2]; } u;
    u.q[0] = *(const int4*)(p);
    u.q[1] = *(const int4*)(p + 16);
    return u.v;
}

// Async global->LDS copy of 16 bytes (per lane), tracked by ASYNCcnt.
// No VGPR staging; LDS dest address is the low 32 bits of the generic pointer.
__device__ __forceinline__ void async_copy_b128(_Float16* lds_dst,
                                                const _Float16* gsrc) {
    uint32_t l = (uint32_t)(uintptr_t)lds_dst;
    uint64_t g = (uint64_t)(uintptr_t)gsrc;
    asm volatile("global_load_async_to_lds_b128 %0, %1, off"
                 :: "v"(l), "v"(g) : "memory");
}
__device__ __forceinline__ void wait_async_zero() {
    asm volatile("s_wait_asynccnt 0x0" ::: "memory");
}

// ------------------------------------------------------------------
// Phase 1: qkv split + RMSNorm(q,k) + RoPE(q,k), emit f16.
//   q,k -> [H][N][128] row-major ; v -> [H][128][N] (transposed)
// One wave32 handles one (token, head): 4 f32 elems per lane.
// ------------------------------------------------------------------
__global__ __launch_bounds__(256)
void qkv_prep_kernel(const float* __restrict__ x,
                     const float* __restrict__ qw,
                     const float* __restrict__ kw,
                     const float* __restrict__ f1,
                     const float* __restrict__ f2,
                     _Float16* __restrict__ qh,
                     _Float16* __restrict__ kh,
                     _Float16* __restrict__ vt,
                     int N1, int txt_len)
{
    const int wave = threadIdx.x >> 5;
    const int lane = threadIdx.x & 31;
    const int flat = blockIdx.x * 8 + wave;
    const int n = flat / HCNT;
    const int h = flat % HCNT;
    if (n >= N1) return;

    const size_t dim = (size_t)HCNT * DH;
    const float* xb = x + (size_t)n * 3 * dim;

    float4 q = *(const float4*)(xb +           (size_t)h * DH + lane * 4);
    float4 k = *(const float4*)(xb +   dim +   (size_t)h * DH + lane * 4);
    float4 v = *(const float4*)(xb + 2*dim +   (size_t)h * DH + lane * 4);

    // RMS over the 128 dims of this head (wave-wide reduction)
    float sq = q.x*q.x + q.y*q.y + q.z*q.z + q.w*q.w;
    float sk = k.x*k.x + k.y*k.y + k.z*k.z + k.w*k.w;
    #pragma unroll
    for (int m = 16; m >= 1; m >>= 1) {
        sq += __shfl_xor(sq, m, 32);
        sk += __shfl_xor(sk, m, 32);
    }
    const float rq = rsqrtf(sq * (1.0f / DH) + 1e-5f);
    const float rk = rsqrtf(sk * (1.0f / DH) + 1e-5f);

    float4 wq = *(const float4*)(qw + lane * 4);
    float4 wk = *(const float4*)(kw + lane * 4);
    q.x *= rq * wq.x; q.y *= rq * wq.y; q.z *= rq * wq.z; q.w *= rq * wq.w;
    k.x *= rk * wk.x; k.y *= rk * wk.y; k.z *= rk * wk.z; k.w *= rk * wk.w;

    // RoPE: this lane holds dims 4L..4L+3 -> complex pairs (2L) and (2L+1)
    const float* fr = (n < txt_len) ? (f1 + (size_t)n * (DH / 2))
                                    : (f2 + (size_t)(n - txt_len) * (DH / 2));
    const float t0 = fr[lane * 2], t1 = fr[lane * 2 + 1];
    float s0, c0, s1, c1;
    __sincosf(t0, &s0, &c0);
    __sincosf(t1, &s1, &c1);
    { float r = q.x, i = q.y; q.x = r*c0 - i*s0; q.y = r*s0 + i*c0; }
    { float r = q.z, i = q.w; q.z = r*c1 - i*s1; q.w = r*s1 + i*c1; }
    { float r = k.x, i = k.y; k.x = r*c0 - i*s0; k.y = r*s0 + i*c0; }
    { float r = k.z, i = k.w; k.z = r*c1 - i*s1; k.w = r*s1 + i*c1; }

    union H4 { _Float16 h[4]; unsigned long long u; } qo, ko;
    qo.h[0] = (_Float16)q.x; qo.h[1] = (_Float16)q.y;
    qo.h[2] = (_Float16)q.z; qo.h[3] = (_Float16)q.w;
    ko.h[0] = (_Float16)k.x; ko.h[1] = (_Float16)k.y;
    ko.h[2] = (_Float16)k.z; ko.h[3] = (_Float16)k.w;

    const size_t rowoff = ((size_t)h * N1 + n) * DH + lane * 4;
    *(unsigned long long*)(qh + rowoff) = qo.u;
    *(unsigned long long*)(kh + rowoff) = ko.u;

    // V transposed: vt[h][d][n]
    const size_t vb = (size_t)h * DH * (size_t)N1;
    vt[vb + (size_t)(lane*4 + 0) * N1 + n] = (_Float16)v.x;
    vt[vb + (size_t)(lane*4 + 1) * N1 + n] = (_Float16)v.y;
    vt[vb + (size_t)(lane*4 + 2) * N1 + n] = (_Float16)v.z;
    vt[vb + (size_t)(lane*4 + 3) * N1 + n] = (_Float16)v.w;
}

// ------------------------------------------------------------------
// Phase 2: flash attention. grid = (N1/BQ, H), 128 threads = 4 waves.
// Double-buffered K/V tiles fetched with GLOBAL_LOAD_ASYNC_TO_LDS
// (ASYNCcnt) overlapped with WMMA compute; one barrier per K-block.
// ------------------------------------------------------------------
__global__ __launch_bounds__(128)
void flash_attn_kernel(const _Float16* __restrict__ qh,
                       const _Float16* __restrict__ kh,
                       const _Float16* __restrict__ vt,
                       float* __restrict__ out, int N1)
{
    __shared__ __align__(16) _Float16 Qs[BQ * DH];          // 16 KB
    __shared__ __align__(16) _Float16 Ks[2][BK * DH];       // 32 KB
    __shared__ __align__(16) _Float16 Vts[2][DH * BK];      // 32 KB (dim-major)
    __shared__ __align__(16) _Float16 Ps[4][16 * BK];       //  8 KB (per-wave P)

    const int tid  = threadIdx.x;
    const int wave = tid >> 5;
    const int lane = tid & 31;
    const int m16  = lane & 15;
    const int hb   = lane >> 4;       // lane half (K-striping select)
    const int h    = blockIdx.y;
    const int qb   = blockIdx.x;

    // softmax works on scale*s ; fold scale and log2(e) into one constant
    const float C = 0.088388347648318447f * 1.4426950408889634f; // D^-1/2 * log2e

    const _Float16* khead = kh + (size_t)h * N1 * DH;  // K rows, row-major
    const _Float16* vhead = vt + (size_t)h * DH * N1;  // V^T rows, dim-major

    // ---- issue async loads of tile 0 (K + V^T), and load Q block ----
    {
        const _Float16* gk = khead;              // kb = 0
        _Float16* lk = Ks[0];
        _Float16* lv = Vts[0];
        #pragma unroll
        for (int i = 0; i < 8; i++) {
            const int idx = tid + i * 128;       // 0..1023 int4 chunks
            async_copy_b128(lk + idx * 8, gk + idx * 8);
            const int d = idx >> 3, j = idx & 7;
            async_copy_b128(lv + idx * 8, vhead + (size_t)d * N1 + j * 8);
        }
        const int4* src = (const int4*)(qh + ((size_t)h * N1 + (size_t)qb * BQ) * DH);
        int4* dst = (int4*)Qs;
        #pragma unroll
        for (int i = 0; i < 8; i++) dst[tid + i * 128] = src[tid + i * 128];
    }
    wait_async_zero();
    __syncthreads();

    // ---- hoist this wave's Q A-fragments (4 K-chunks of 32) ----
    v16h qfrag[4];
    #pragma unroll
    for (int kc = 0; kc < 4; kc++)
        qfrag[kc] = load_frag(Qs + (wave * 16 + m16) * DH + kc * 32 + hb * 8);

    v8f oacc[8];
    #pragma unroll
    for (int t = 0; t < 8; t++) oacc[t] = {};
    float mrow[8], lrow[8];
    #pragma unroll
    for (int g = 0; g < 8; g++) { mrow[g] = -__builtin_inff(); lrow[g] = 0.0f; }

    const int nkb = N1 / BK;
    for (int kb = 0; kb < nkb; kb++) {
        const int cur = kb & 1;

        // ---- prefetch next K/V tile into the other buffer (async) ----
        if (kb + 1 < nkb) {
            const _Float16* gk = khead + (size_t)(kb + 1) * BK * DH;
            const _Float16* gv = vhead + (size_t)(kb + 1) * BK;
            _Float16* lk = Ks[cur ^ 1];
            _Float16* lv = Vts[cur ^ 1];
            #pragma unroll
            for (int i = 0; i < 8; i++) {
                const int idx = tid + i * 128;
                async_copy_b128(lk + idx * 8, gk + idx * 8);
                const int d = idx >> 3, j = idx & 7;
                async_copy_b128(lv + idx * 8, gv + (size_t)d * N1 + j * 8);
            }
        }

        const _Float16* Kc = Ks[cur];
        const _Float16* Vc = Vts[cur];

        // ---- S(16x64) = Q(16x128) * K^T ----
        v8f sacc[4];
        #pragma unroll
        for (int t = 0; t < 4; t++) sacc[t] = {};
        #pragma unroll
        for (int kc = 0; kc < 4; kc++) {
            #pragma unroll
            for (int nt = 0; nt < 4; nt++) {
                v16h b = load_frag(Kc + (nt * 16 + m16) * DH + kc * 32 + hb * 8);
                sacc[nt] = __builtin_amdgcn_wmma_f32_16x16x32_f16(
                    false, qfrag[kc], false, b, (short)0, sacc[nt], false, false);
            }
        }

        // ---- online softmax: row max across 64 cols (16-lane reduce) ----
        float corr[8];
        #pragma unroll
        for (int g = 0; g < 8; g++) {
            float m0 = fmaxf(fmaxf(sacc[0][g], sacc[1][g]),
                             fmaxf(sacc[2][g], sacc[3][g]));
            m0 = fmaxf(m0, __shfl_xor(m0, 1, 32));
            m0 = fmaxf(m0, __shfl_xor(m0, 2, 32));
            m0 = fmaxf(m0, __shfl_xor(m0, 4, 32));
            m0 = fmaxf(m0, __shfl_xor(m0, 8, 32));
            const float mn = fmaxf(mrow[g], m0);
            corr[g] = __builtin_exp2f((mrow[g] - mn) * C);
            mrow[g] = mn;
        }

        // ---- P = exp2((s-m)*C), row sums, spill P(f16) to LDS ----
        float rs[8];
        #pragma unroll
        for (int g = 0; g < 8; g++) rs[g] = 0.0f;
        _Float16* pw = Ps[wave];
        #pragma unroll
        for (int nt = 0; nt < 4; nt++) {
            #pragma unroll
            for (int g = 0; g < 8; g++) {
                const float p = __builtin_exp2f((sacc[nt][g] - mrow[g]) * C);
                rs[g] += p;
                pw[(g + 8 * hb) * BK + nt * 16 + m16] = (_Float16)p;
            }
        }
        #pragma unroll
        for (int g = 0; g < 8; g++) {
            float s = rs[g];
            s += __shfl_xor(s, 1, 32);
            s += __shfl_xor(s, 2, 32);
            s += __shfl_xor(s, 4, 32);
            s += __shfl_xor(s, 8, 32);
            lrow[g] = lrow[g] * corr[g] + s;
        }
        #pragma unroll
        for (int nt = 0; nt < 8; nt++)
            #pragma unroll
            for (int g = 0; g < 8; g++)
                oacc[nt][g] *= corr[g];

        // ---- O(16x128) += P(16x64) * V(64x128) ----
        // Same-wave LDS store->load is in-order; no barrier needed for Ps.
        #pragma unroll
        for (int kc = 0; kc < 2; kc++) {
            v16h a = load_frag(pw + m16 * BK + kc * 32 + hb * 8);
            #pragma unroll
            for (int nt = 0; nt < 8; nt++) {
                v16h b = load_frag(Vc + (nt * 16 + m16) * BK + kc * 32 + hb * 8);
                oacc[nt] = __builtin_amdgcn_wmma_f32_16x16x32_f16(
                    false, a, false, b, (short)0, oacc[nt], false, false);
            }
        }

        // ---- next tile ready + everyone done with current buffers ----
        wait_async_zero();
        __syncthreads();
    }

    // ---- epilogue: O /= l, scatter to out[N][H*D] fp32 ----
    float inv[8];
    #pragma unroll
    for (int g = 0; g < 8; g++) inv[g] = 1.0f / lrow[g];
    float* ob = out + ((size_t)qb * BQ + wave * 16) * (HCNT * DH) + (size_t)h * DH;
    #pragma unroll
    for (int g = 0; g < 8; g++) {
        float* orow = ob + (size_t)(g + 8 * hb) * (HCNT * DH);
        #pragma unroll
        for (int nt = 0; nt < 8; nt++)
            orow[nt * 16 + m16] = oacc[nt][g] * inv[g];
    }
}

extern "C" void kernel_launch(void* const* d_in, const int* in_sizes, int n_in,
                              void* d_out, int out_size, void* d_ws, size_t ws_size,
                              hipStream_t stream) {
    const float* x  = (const float*)d_in[0];
    const float* qw = (const float*)d_in[1];
    const float* kw = (const float*)d_in[2];
    const float* f1 = (const float*)d_in[3];
    const float* f2 = (const float*)d_in[4];

    const int D       = in_sizes[1];                 // 128
    const int dim     = HCNT * D;                    // 2048
    const int N1      = in_sizes[0] / (3 * dim);     // 4096
    const int txt_len = in_sizes[3] / (D / 2);       // 3072

    float* out = (float*)d_out;

    const size_t per = (size_t)HCNT * N1 * D;        // halves per tensor
    _Float16* qh = (_Float16*)d_ws;
    _Float16* kh = qh + per;
    _Float16* vt = kh + per;

    dim3 g1((N1 * HCNT) / 8);
    qkv_prep_kernel<<<g1, 256, 0, stream>>>(x, qw, kw, f1, f2, qh, kh, vt,
                                            N1, txt_len);

    dim3 g2(N1 / BQ, HCNT);
    flash_attn_kernel<<<g2, 128, 0, stream>>>(qh, kh, vt, out, N1);
}
